// AttLayer_69398081569425
// MI455X (gfx1250) — compile-verified, matrix operand order checked by reading, not run on hardware
//
#include <hip/hip_runtime.h>

// ---------------------------------------------------------------------------
// Problem constants (from reference): B=1, C=2048, L=16384, Cq=1024, BL=512
// ---------------------------------------------------------------------------
#define C_DIM   2048
#define L_DIM   16384
#define CQ_DIM  1024
#define NB      32
#define BL_DIM  512
#define WIN     1024          // 2*BL window
#define HALF    256           // BL/2 padding

typedef unsigned short bf16_t;
typedef __attribute__((ext_vector_type(16))) __bf16 v16bf;
typedef __attribute__((ext_vector_type(8)))  float  v8f;

// ---------------------------------------------------------------------------
// gfx1250 async global->LDS copy (ASYNCcnt-tracked), guarded by __has_builtin
// ---------------------------------------------------------------------------
#if defined(__has_builtin)
#if __has_builtin(__builtin_amdgcn_global_load_async_to_lds_b128)
#define HAVE_ASYNC_LDS 1
#endif
#endif

#ifdef HAVE_ASYNC_LDS
// builtin expects: (v4i __device__* src, v4i __shared__* dst, imm off, imm cpol)
typedef int v4i_t __attribute__((vector_size(16)));
typedef __attribute__((address_space(1))) v4i_t g_v4i;
typedef __attribute__((address_space(3))) v4i_t l_v4i;

__device__ __forceinline__ void async_copy_b128(const void* g, void* l) {
  // generic global ptr == 64-bit global address; generic LDS ptr's low dword
  // is the wave-relative LDS byte address (aperture encoding).
  __builtin_amdgcn_global_load_async_to_lds_b128(
      (g_v4i*)(unsigned long long)g,
      (l_v4i*)(unsigned)(unsigned long long)l, 0, 0);
}
#endif

__device__ __forceinline__ void async_wait_all() {
#if defined(__has_builtin)
#if __has_builtin(__builtin_amdgcn_s_wait_asynccnt)
  __builtin_amdgcn_s_wait_asynccnt(0);
#endif
#endif
}

// ---------------------------------------------------------------------------
// helpers
// ---------------------------------------------------------------------------
__device__ __forceinline__ bf16_t f2bf(float f) {
  union { float f; unsigned u; } c; c.f = f;
  unsigned r = c.u + 0x7FFFu + ((c.u >> 16) & 1u);
  return (bf16_t)(r >> 16);
}

__device__ __forceinline__ v8f vzero8() {
  v8f z;
#pragma unroll
  for (int i = 0; i < 8; ++i) z[i] = 0.0f;
  return z;
}

union FragBF { uint4 u[2]; v16bf v; };

// Read a 16x32 bf16 A/B fragment from an LDS tile laid out [128][32].
// Lanes 0-15 carry rows r with K=0..15, lanes 16-31 carry rows r with K=16..31.
__device__ __forceinline__ v16bf load_frag(const bf16_t* __restrict__ tile,
                                           int row_base, int lane) {
  FragBF f;
  int r  = row_base + (lane & 15);
  int kh = (lane >> 4) * 16;
  const uint4* p = (const uint4*)(tile + r * 32 + kh);
  f.u[0] = p[0];
  f.u[1] = p[1];
  return f.v;
}

// Stage a 128(row) x 32(col) tile from row-major src into dst[128][32].
// col0 may be negative (window padding); out-of-range chunks are zero-filled.
// Uses CDNA5 async global->LDS B128 copies when available.
__device__ __forceinline__ void stage_rowmajor_128x32(
    bf16_t* __restrict__ dst, const bf16_t* __restrict__ src,
    int row0, long col0, int ld, long colLimit, int tid) {
#pragma unroll
  for (int it = 0; it < 2; ++it) {
    int idx = tid * 2 + it;          // 0..511 chunks of 8 bf16
    int row = idx >> 2;              // 0..127
    int ck  = (idx & 3) * 8;         // 0,8,16,24
    long gc = col0 + ck;
    bf16_t* ldst = &dst[row * 32 + ck];
#ifdef HAVE_ASYNC_LDS
    if (gc >= 0 && gc + 8 <= colLimit) {
      async_copy_b128(src + (long)(row0 + row) * ld + gc, ldst);
    } else {
      *(uint4*)ldst = make_uint4(0u, 0u, 0u, 0u);
    }
#else
    uint4 val = make_uint4(0u, 0u, 0u, 0u);
    if (gc >= 0 && gc + 8 <= colLimit)
      val = *(const uint4*)(src + (long)(row0 + row) * ld + gc);
    *(uint4*)ldst = val;
#endif
  }
}

// Stage a 32(row) x 128(col) region of row-major src, transposed, into
// dst[128][32]:  dst[x][y] = src[row0+y][col0+x].  Zero-fill out of range.
__device__ __forceinline__ void stage_transpose_32x128(
    bf16_t* __restrict__ dst, const bf16_t* __restrict__ src,
    int row0, long col0, int ld, long colLimit, int tid) {
#pragma unroll
  for (int it = 0; it < 2; ++it) {
    int idx = tid * 2 + it;          // 0..511
    int r   = idx >> 4;              // 0..31  src row in tile
    int cc  = (idx & 15) * 8;        // 0..120 src col in tile
    long gc = col0 + cc;
    bf16_t tmp[8];
    if (gc >= 0 && gc + 8 <= colLimit) {
      uint4 v = *(const uint4*)(src + (long)(row0 + r) * ld + gc);
      *(uint4*)tmp = v;
    } else {
#pragma unroll
      for (int j = 0; j < 8; ++j) tmp[j] = 0;
    }
#pragma unroll
    for (int j = 0; j < 8; ++j)
      dst[(cc + j) * 32 + r] = tmp[j];
  }
}

// One K=32 step for a wave's 64x32 output tile: 4 A frags x 2 B frags = 8 WMMA.
__device__ __forceinline__ void wave_mma_step(const bf16_t* __restrict__ As,
                                              const bf16_t* __restrict__ Bs,
                                              int wm, int wn, int lane,
                                              v8f acc[4][2]) {
  v16bf b0 = load_frag(Bs, wn * 32 + 0,  lane);
  v16bf b1 = load_frag(Bs, wn * 32 + 16, lane);
#pragma unroll
  for (int mt = 0; mt < 4; ++mt) {
    v16bf a = load_frag(As, wm * 64 + mt * 16, lane);
    acc[mt][0] = __builtin_amdgcn_wmma_f32_16x16x32_bf16(
        false, a, false, b0, (short)0, acc[mt][0], false, false);
    acc[mt][1] = __builtin_amdgcn_wmma_f32_16x16x32_bf16(
        false, a, false, b1, (short)0, acc[mt][1], false, false);
  }
}

// ---------------------------------------------------------------------------
// conversion kernels
// ---------------------------------------------------------------------------
__global__ void k_cvt_f32_bf16(const float* __restrict__ s,
                               bf16_t* __restrict__ d, long n) {
  long i = (long)blockIdx.x * blockDim.x + threadIdx.x;
  long st = (long)gridDim.x * blockDim.x;
  for (; i < n; i += st) d[i] = f2bf(s[i]);
}

__global__ void k_pack_wqkv(const float* __restrict__ wq,
                            const float* __restrict__ wk,
                            const float* __restrict__ wv,
                            bf16_t* __restrict__ d) {
  long n = (long)(3 * CQ_DIM) * C_DIM;
  long i = (long)blockIdx.x * blockDim.x + threadIdx.x;
  long st = (long)gridDim.x * blockDim.x;
  for (; i < n; i += st) {
    int row = (int)(i / C_DIM);
    int col = (int)(i % C_DIM);
    const float* src = (row < CQ_DIM) ? (wq + (long)row * C_DIM)
                     : (row < 2 * CQ_DIM) ? (wk + (long)(row - CQ_DIM) * C_DIM)
                     : (wv + (long)(row - 2 * CQ_DIM) * C_DIM);
    d[i] = f2bf(src[col]);
  }
}

// ---------------------------------------------------------------------------
// GEMM 1: QKV projection.  C[3072 x 16384] = Wqkv[3072 x 2048] * X[2048 x 16384]
// Double-buffered: async copies for step i+1 overlap WMMA on step i.
// ---------------------------------------------------------------------------
__global__ __launch_bounds__(256) void k_gemm_qkv(
    const bf16_t* __restrict__ A, const bf16_t* __restrict__ B,
    bf16_t* __restrict__ out,
    const float* __restrict__ bq, const float* __restrict__ bk,
    const float* __restrict__ bv) {
  __shared__ __align__(16) bf16_t As[2][128 * 32];
  __shared__ __align__(16) bf16_t Bs[2][128 * 32];
  const int tid = threadIdx.x, lane = tid & 31, wave = tid >> 5;
  const int wm = wave >> 2, wn = wave & 3;
  const int m0 = blockIdx.y * 128, n0 = blockIdx.x * 128;
  const int NSTEP = C_DIM / 32;

  v8f acc[4][2];
#pragma unroll
  for (int i = 0; i < 4; ++i) { acc[i][0] = vzero8(); acc[i][1] = vzero8(); }

  stage_rowmajor_128x32(As[0], A, m0, 0, C_DIM, C_DIM, tid);
  stage_transpose_32x128(Bs[0], B, 0, n0, L_DIM, (long)L_DIM, tid);

  for (int i = 0; i < NSTEP; ++i) {
    async_wait_all();
    __syncthreads();
    if (i + 1 < NSTEP) {
      int k1 = (i + 1) * 32;
      int nb = (i + 1) & 1;
      stage_rowmajor_128x32(As[nb], A, m0, k1, C_DIM, C_DIM, tid);
      stage_transpose_32x128(Bs[nb], B, k1, n0, L_DIM, (long)L_DIM, tid);
      __builtin_prefetch(B + (long)(k1 + 32 + (tid >> 3)) * L_DIM + n0, 0, 0);
    }
    wave_mma_step(As[i & 1], Bs[i & 1], wm, wn, lane, acc);
  }

  const int half = lane >> 4, lc = lane & 15;
#pragma unroll
  for (int mt = 0; mt < 4; ++mt)
#pragma unroll
    for (int nt = 0; nt < 2; ++nt) {
      int rowb = m0 + wm * 64 + mt * 16 + half * 8;
      int col  = n0 + wn * 32 + nt * 16 + lc;
#pragma unroll
      for (int r = 0; r < 8; ++r) {
        int row = rowb + r;
        float bias = (row < CQ_DIM) ? bq[row]
                   : (row < 2 * CQ_DIM) ? bk[row - CQ_DIM]
                   : bv[row - 2 * CQ_DIM];
        out[(long)row * L_DIM + col] = f2bf(acc[mt][nt][r] + bias);
      }
    }
}

// ---------------------------------------------------------------------------
// GEMM 2: energy per block.  E[l,m] = (1/32) * sum_c q[c, n*512+l] * k[c, gm]
//   gm = n*512 + m - 256, zero outside [0, L).   M=512, N=1024, K=1024.
// ---------------------------------------------------------------------------
__global__ __launch_bounds__(256) void k_gemm_energy(
    const bf16_t* __restrict__ q, const bf16_t* __restrict__ kk,
    float* __restrict__ E) {
  __shared__ __align__(16) bf16_t As[2][128 * 32];
  __shared__ __align__(16) bf16_t Bs[2][128 * 32];
  const int tid = threadIdx.x, lane = tid & 31, wave = tid >> 5;
  const int wm = wave >> 2, wn = wave & 3;
  const int nblk = blockIdx.z;
  const int l0 = blockIdx.y * 128;               // 0..384
  const int m0 = blockIdx.x * 128;               // 0..896
  const long lcol0 = (long)nblk * BL_DIM + l0;   // q column base
  const long mcol0 = (long)nblk * BL_DIM - HALF + m0;  // k column base (padded)
  const int NSTEP = CQ_DIM / 32;

  v8f acc[4][2];
#pragma unroll
  for (int i = 0; i < 4; ++i) { acc[i][0] = vzero8(); acc[i][1] = vzero8(); }

  stage_transpose_32x128(As[0], q,  0, lcol0, L_DIM, (long)L_DIM, tid);
  stage_transpose_32x128(Bs[0], kk, 0, mcol0, L_DIM, (long)L_DIM, tid);

  for (int i = 0; i < NSTEP; ++i) {
    __syncthreads();
    if (i + 1 < NSTEP) {
      int k1 = (i + 1) * 32;
      int nb = (i + 1) & 1;
      stage_transpose_32x128(As[nb], q,  k1, lcol0, L_DIM, (long)L_DIM, tid);
      stage_transpose_32x128(Bs[nb], kk, k1, mcol0, L_DIM, (long)L_DIM, tid);
    }
    wave_mma_step(As[i & 1], Bs[i & 1], wm, wn, lane, acc);
  }

  const float scale = 0.03125f;   // 1/sqrt(1024)
  const int half = lane >> 4, lc = lane & 15;
#pragma unroll
  for (int mt = 0; mt < 4; ++mt)
#pragma unroll
    for (int nt = 0; nt < 2; ++nt) {
      int l = l0 + wm * 64 + mt * 16 + half * 8;
      int m = m0 + wn * 32 + nt * 16 + lc;
#pragma unroll
      for (int r = 0; r < 8; ++r)
        E[((long)nblk * BL_DIM + (l + r)) * WIN + m] = acc[mt][nt][r] * scale;
    }
}

// ---------------------------------------------------------------------------
// softmax: one wave32 per (block, l) row of 1024 energies.
// final_mask = window(m) * padded_mask;  attn stored TRANSPOSED (bf16):
// attnT[n][m][l]  so the AV gemm reads its B operand in natural layout.
// ---------------------------------------------------------------------------
__global__ __launch_bounds__(256) void k_softmax(
    const float* __restrict__ E, const float* __restrict__ gmask,
    bf16_t* __restrict__ attnT) {
  const int lane = threadIdx.x & 31;
  const int wave = threadIdx.x >> 5;
  const long row = (long)blockIdx.x * 8 + wave;  // 0 .. 16383
  const int nblk = (int)(row >> 9);
  const int l    = (int)(row & 511);
  const float* e = E + row * WIN;

  float vals[32], mk[32];
  float mx = -1e30f;
#pragma unroll
  for (int i = 0; i < 32; ++i) {
    int m = lane + i * 32;
    long gcol = (long)nblk * BL_DIM + m - HALF;
    float pm = (gcol >= 0 && gcol < L_DIM) ? gmask[gcol] : 0.0f;
    float wm = (m < WIN - 1) ? 1.0f : 0.0f;  // reference window-mask semantics
    float fm = pm * wm;
    mk[i] = fm;
    float ev = e[m] + __logf(fm + 1e-9f);
    vals[i] = ev;
    mx = fmaxf(mx, ev);
  }
#pragma unroll
  for (int off = 16; off; off >>= 1) mx = fmaxf(mx, __shfl_xor(mx, off, 32));
  float sum = 0.0f;
#pragma unroll
  for (int i = 0; i < 32; ++i) { vals[i] = __expf(vals[i] - mx); sum += vals[i]; }
#pragma unroll
  for (int off = 16; off; off >>= 1) sum += __shfl_xor(sum, off, 32);
  float inv = 1.0f / sum;
#pragma unroll
  for (int i = 0; i < 32; ++i) {
    int m = lane + i * 32;
    attnT[(long)nblk * WIN * BL_DIM + (long)m * BL_DIM + l] =
        f2bf(vals[i] * inv * mk[i]);
  }
}

// ---------------------------------------------------------------------------
// GEMM 3: AV per block.  mid[c, n*512+l] = relu( sum_m v[c, gm] * attnT[m, l] )
//   M=1024 (c), N=512 (l), K=1024 (m, padded on the v side).
// ---------------------------------------------------------------------------
__global__ __launch_bounds__(256) void k_gemm_av(
    const bf16_t* __restrict__ v, const bf16_t* __restrict__ attnT,
    bf16_t* __restrict__ mid) {
  __shared__ __align__(16) bf16_t As[2][128 * 32];
  __shared__ __align__(16) bf16_t Bs[2][128 * 32];
  const int tid = threadIdx.x, lane = tid & 31, wave = tid >> 5;
  const int wm = wave >> 2, wn = wave & 3;
  const int nblk = blockIdx.z;
  const int c0 = blockIdx.y * 128;
  const int l0 = blockIdx.x * 128;
  const bf16_t* Bblk = attnT + (long)nblk * WIN * BL_DIM;
  const long gmbase = (long)nblk * BL_DIM - HALF;  // padded v column base
  const int NSTEP = WIN / 32;

  v8f acc[4][2];
#pragma unroll
  for (int i = 0; i < 4; ++i) { acc[i][0] = vzero8(); acc[i][1] = vzero8(); }

  stage_rowmajor_128x32(As[0], v, c0, gmbase, L_DIM, (long)L_DIM, tid);
  stage_transpose_32x128(Bs[0], Bblk, 0, (long)l0, BL_DIM, (long)BL_DIM, tid);

  for (int i = 0; i < NSTEP; ++i) {
    async_wait_all();
    __syncthreads();
    if (i + 1 < NSTEP) {
      int k1 = (i + 1) * 32;
      int nb = (i + 1) & 1;
      stage_rowmajor_128x32(As[nb], v, c0, gmbase + k1, L_DIM, (long)L_DIM, tid);
      stage_transpose_32x128(Bs[nb], Bblk, k1, (long)l0, BL_DIM, (long)BL_DIM, tid);
    }
    wave_mma_step(As[i & 1], Bs[i & 1], wm, wn, lane, acc);
  }

  const int half = lane >> 4, lc = lane & 15;
#pragma unroll
  for (int mt = 0; mt < 4; ++mt)
#pragma unroll
    for (int nt = 0; nt < 2; ++nt) {
      int c = c0 + wm * 64 + mt * 16 + half * 8;
      int l = l0 + wn * 32 + nt * 16 + lc;
#pragma unroll
      for (int r = 0; r < 8; ++r) {
        float x = acc[mt][nt][r];
        x = x > 0.0f ? x : 0.0f;                         // relu
        mid[(long)(c + r) * L_DIM + (long)nblk * BL_DIM + l] = f2bf(x);
      }
    }
}

// ---------------------------------------------------------------------------
// GEMM 4: output projection.  out[o,l] = (Wo[o,:] . mid[:,l] + bo[o]) * mask[l]
//   M=2048, N=16384, K=1024.  fp32 output.
// ---------------------------------------------------------------------------
__global__ __launch_bounds__(256) void k_gemm_out(
    const bf16_t* __restrict__ A, const bf16_t* __restrict__ B,
    const float* __restrict__ bo, const float* __restrict__ gmask,
    float* __restrict__ out) {
  __shared__ __align__(16) bf16_t As[2][128 * 32];
  __shared__ __align__(16) bf16_t Bs[2][128 * 32];
  const int tid = threadIdx.x, lane = tid & 31, wave = tid >> 5;
  const int wm = wave >> 2, wn = wave & 3;
  const int m0 = blockIdx.y * 128, n0 = blockIdx.x * 128;
  const int NSTEP = CQ_DIM / 32;

  v8f acc[4][2];
#pragma unroll
  for (int i = 0; i < 4; ++i) { acc[i][0] = vzero8(); acc[i][1] = vzero8(); }

  stage_rowmajor_128x32(As[0], A, m0, 0, CQ_DIM, CQ_DIM, tid);
  stage_transpose_32x128(Bs[0], B, 0, n0, L_DIM, (long)L_DIM, tid);

  for (int i = 0; i < NSTEP; ++i) {
    async_wait_all();
    __syncthreads();
    if (i + 1 < NSTEP) {
      int k1 = (i + 1) * 32;
      int nb = (i + 1) & 1;
      stage_rowmajor_128x32(As[nb], A, m0, k1, CQ_DIM, CQ_DIM, tid);
      stage_transpose_32x128(Bs[nb], B, k1, n0, L_DIM, (long)L_DIM, tid);
      __builtin_prefetch(B + (long)(k1 + 32 + (tid >> 3)) * L_DIM + n0, 0, 0);
    }
    wave_mma_step(As[i & 1], Bs[i & 1], wm, wn, lane, acc);
  }

  const int half = lane >> 4, lc = lane & 15;
#pragma unroll
  for (int mt = 0; mt < 4; ++mt)
#pragma unroll
    for (int nt = 0; nt < 2; ++nt) {
      int o = m0 + wm * 64 + mt * 16 + half * 8;
      int l = n0 + wn * 32 + nt * 16 + lc;
      float mv = gmask[l];
#pragma unroll
      for (int r = 0; r < 8; ++r)
        out[(long)(o + r) * L_DIM + l] = (acc[mt][nt][r] + bo[o + r]) * mv;
    }
}

// ---------------------------------------------------------------------------
// host launcher
// ---------------------------------------------------------------------------
extern "C" void kernel_launch(void* const* d_in, const int* in_sizes, int n_in,
                              void* d_out, int out_size, void* d_ws,
                              size_t ws_size, hipStream_t stream) {
  (void)in_sizes; (void)n_in; (void)out_size; (void)ws_size;
  const float* x1  = (const float*)d_in[0];
  // d_in[1] (x2) unused by the reference
  const float* msk = (const float*)d_in[2];
  const float* wq  = (const float*)d_in[3];
  const float* bq  = (const float*)d_in[4];
  const float* wk  = (const float*)d_in[5];
  const float* bk  = (const float*)d_in[6];
  const float* wv  = (const float*)d_in[7];
  const float* bv  = (const float*)d_in[8];
  const float* wo  = (const float*)d_in[9];
  const float* bo  = (const float*)d_in[10];
  float* out = (float*)d_out;

  char* ws = (char*)d_ws;
  // workspace layout (bytes); energy reuses the x1-bf16 region (dead by then)
  const size_t o_x1b  = 0;                              // 2048*16384*2 = 64 MB
  const size_t o_enrg = 0;                              // 32*512*1024*4 = 64 MB
  const size_t o_wqkv = (size_t)C_DIM * L_DIM * 2;      // +12 MB
  const size_t o_wob  = o_wqkv + (size_t)3 * CQ_DIM * C_DIM * 2;  // +4 MB
  const size_t o_qkv  = o_wob  + (size_t)C_DIM * CQ_DIM * 2;      // +96 MB
  const size_t o_attn = o_qkv  + (size_t)3 * CQ_DIM * L_DIM * 2;  // +32 MB
  const size_t o_mid  = o_attn + (size_t)NB * WIN * BL_DIM * 2;   // +32 MB

  bf16_t* x1b   = (bf16_t*)(ws + o_x1b);
  float*  enrg  = (float*) (ws + o_enrg);
  bf16_t* wqkvb = (bf16_t*)(ws + o_wqkv);
  bf16_t* wob   = (bf16_t*)(ws + o_wob);
  bf16_t* qkvb  = (bf16_t*)(ws + o_qkv);
  bf16_t* attnT = (bf16_t*)(ws + o_attn);
  bf16_t* midb  = (bf16_t*)(ws + o_mid);

  // 1. convert inputs/weights to bf16
  k_cvt_f32_bf16<<<8192, 256, 0, stream>>>(x1, x1b, (long)C_DIM * L_DIM);
  k_pack_wqkv<<<2048, 256, 0, stream>>>(wq, wk, wv, wqkvb);
  k_cvt_f32_bf16<<<1024, 256, 0, stream>>>(wo, wob, (long)C_DIM * CQ_DIM);

  // 2. fused QKV projection: [3072 x 16384]
  k_gemm_qkv<<<dim3(L_DIM / 128, (3 * CQ_DIM) / 128), 256, 0, stream>>>(
      wqkvb, x1b, qkvb, bq, bk, bv);

  const bf16_t* qb = qkvb;
  const bf16_t* kb = qkvb + (size_t)CQ_DIM * L_DIM;
  const bf16_t* vb = qkvb + (size_t)2 * CQ_DIM * L_DIM;

  // 3. block energies: 32 blocks x [512 x 1024]
  k_gemm_energy<<<dim3(WIN / 128, BL_DIM / 128, NB), 256, 0, stream>>>(
      qb, kb, enrg);

  // 4. masked softmax -> attn^T (bf16)
  k_softmax<<<(NB * BL_DIM) / 8, 256, 0, stream>>>(enrg, msk, attnT);

  // 5. AV: 32 blocks x [1024 x 512], relu -> mid
  k_gemm_av<<<dim3(BL_DIM / 128, CQ_DIM / 128, NB), 256, 0, stream>>>(
      vb, attnT, midb);

  // 6. output projection + mask
  k_gemm_out<<<dim3(L_DIM / 128, C_DIM / 128), 256, 0, stream>>>(
      wob, midb, bo, msk, out);
}
